// SqueezedGRU_S_4990751998018
// MI455X (gfx1250) — matrix operand order repaired and checked by reading, not compiled, not run on hardware
//
#include <hip/hip_runtime.h>
#include <stdint.h>

typedef __bf16 bf16;
typedef __attribute__((ext_vector_type(16))) __bf16 v16bf;
typedef __attribute__((ext_vector_type(8)))  float  v8f;

#define BATCH 16384
#define HDIM  512
#define NGROUP 8
#define GSZ    64
#define GATES  1536   // 3*HDIM

static __device__ __forceinline__ bf16 f2bf(float f) {
  union { float f; uint32_t u; } v; v.f = f;
  uint32_t r = v.u + 0x7FFFu + ((v.u >> 16) & 1u);   // round-to-nearest-even
  union { uint16_t s; bf16 b; } o; o.s = (uint16_t)(r >> 16);
  return o.b;
}

union Frag { v16bf v; uint4 q[2]; bf16 h[16]; };

// A fragment (16-bit 16x32, ISA 7.12.2): lane m=l&15, sel=l>>4.
// VGPR0-3 hold K = kb+sel*8 .. +7 ; VGPR4-7 hold K = kb+16+sel*8 .. +7.
static __device__ __forceinline__ v16bf load_a(const bf16* p, int sel) {
  Frag f;
  f.q[0] = *(const uint4*)(p + sel * 8);
  f.q[1] = *(const uint4*)(p + 16 + sel * 8);
  return f.v;
}
static __device__ __forceinline__ v16bf load_a(const float* p, int sel) {
  float lo[8], hi[8];
  *(float4*)(lo)     = *(const float4*)(p + sel * 8);
  *(float4*)(lo + 4) = *(const float4*)(p + sel * 8 + 4);
  *(float4*)(hi)     = *(const float4*)(p + 16 + sel * 8);
  *(float4*)(hi + 4) = *(const float4*)(p + 16 + sel * 8 + 4);
  Frag f;
#pragma unroll
  for (int e = 0; e < 8; ++e) { f.h[e] = f2bf(lo[e]); f.h[8 + e] = f2bf(hi[e]); }
  return f.v;
}
// B fragment (32x16): lane n holds column n; VGPR0-3 K=kb+sel*8..+7, VGPR4-7 K=kb+16+sel*8..+7,
// read from the (pre-transposed) weight row.
static __device__ __forceinline__ v16bf load_b(const bf16* p, int sel) {
  Frag f;
  f.q[0] = *(const uint4*)(p + sel * 8);
  f.q[1] = *(const uint4*)(p + 16 + sel * 8);
  return f.v;
}

static __device__ __forceinline__ v8f wmma_bf16(v16bf a, v16bf b, v8f c) {
  return __builtin_amdgcn_wmma_f32_16x16x32_bf16(false, a, false, b, (short)0, c, false, false);
}

static __device__ __forceinline__ void store_c(float* p, float x) { *p = x; }
static __device__ __forceinline__ void store_c(bf16* p, float x) { *p = f2bf(x); }

// ---------------------------------------------------------------------------
// Weight conversion: w_in/w_out transposed to [g][out][in] bf16; W_ih straight copy.
__global__ void k_convert(const float* __restrict__ w_in, const float* __restrict__ W_ih,
                          const float* __restrict__ w_out,
                          bf16* __restrict__ wInT, bf16* __restrict__ wOutT,
                          bf16* __restrict__ WihB) {
  const int nGrp = NGROUP * GSZ * GSZ;            // 32768
  const int total = 2 * nGrp + 2 * GATES * HDIM;  // + 1,572,864
  for (int i = blockIdx.x * blockDim.x + threadIdx.x; i < total;
       i += gridDim.x * blockDim.x) {
    if (i < nGrp) {
      int g = i >> 12, r = i & 4095, h = r >> 6, ii = r & 63;
      wInT[i] = f2bf(w_in[(g << 12) + (ii << 6) + h]);
    } else if (i < 2 * nGrp) {
      int j = i - nGrp;
      int g = j >> 12, r = j & 4095, h = r >> 6, ii = r & 63;
      wOutT[j] = f2bf(w_out[(g << 12) + (ii << 6) + h]);
    } else {
      int j = i - 2 * nGrp;
      WihB[j] = f2bf(W_ih[j]);
    }
  }
}

// ---------------------------------------------------------------------------
// Grouped (block-diagonal) projection: per-wave 16x16 tile, K = 64.
template <typename AT, typename OT>
__global__ void k_grouped(const AT* __restrict__ X, const bf16* __restrict__ WT,
                          const float* __restrict__ bias, OT* __restrict__ Y) {
  const int lane = threadIdx.x & 31;
  const int wave = threadIdx.x >> 5;
  const int t = blockIdx.x * 4 + wave;      // [0, 32768)
  const int rt = t >> 5, ct = t & 31;
  const int row0 = rt * 16;
  const int g = ct >> 2;
  const int c0g = (ct & 3) * 16;
  const int m = lane & 15, sel = lane >> 4;

  const AT*   arow = X + (size_t)(row0 + m) * HDIM + g * GSZ;
  const bf16* brow = WT + (g << 12) + (c0g + m) * GSZ;

  v8f c = {};
#pragma unroll
  for (int kb = 0; kb < GSZ; kb += 32) {
    v16bf a = load_a(arow + kb, sel);
    v16bf b = load_b(brow + kb, sel);
    c = wmma_bf16(a, b, c);
  }
  const int col = g * GSZ + c0g + m;
  const float bs = bias[col];
#pragma unroll
  for (int v = 0; v < 8; ++v) {
    const int row = row0 + v + 8 * sel;
    store_c(Y + (size_t)row * HDIM + col, c[v] + bs);
  }
}

// ---------------------------------------------------------------------------
// GRU layer with h_prev == 0:  gi = X @ Wih^T ; r=sig(ir+bihr+bhhr), z=sig(...),
// n=tanh(inn+bihn + r*bhhn), h' = (1-z)*n.  3 gate accumulators share one A frag.
__global__ void k_gru(const bf16* __restrict__ X, const bf16* __restrict__ Wih,
                      const float* __restrict__ bih, const float* __restrict__ bhh,
                      bf16* __restrict__ Hout) {
  const int lane = threadIdx.x & 31;
  const int wave = threadIdx.x >> 5;
  const int t = blockIdx.x * 4 + wave;
  const int rt = t >> 5, ct = t & 31;
  const int row0 = rt * 16;
  const int col0 = ct * 16;
  const int m = lane & 15, sel = lane >> 4;

  const bf16* arow = X + (size_t)(row0 + m) * HDIM;
  const bf16* wr = Wih + (size_t)(0 * HDIM + col0 + m) * HDIM;
  const bf16* wz = Wih + (size_t)(1 * HDIM + col0 + m) * HDIM;
  const bf16* wn = Wih + (size_t)(2 * HDIM + col0 + m) * HDIM;

  v8f cr = {}, cz = {}, cn = {};
#pragma unroll 4
  for (int kb = 0; kb < HDIM; kb += 32) {
    v16bf a = load_a(arow + kb, sel);
    cr = wmma_bf16(a, load_b(wr + kb, sel), cr);
    cz = wmma_bf16(a, load_b(wz + kb, sel), cz);
    cn = wmma_bf16(a, load_b(wn + kb, sel), cn);
  }

  const int col = col0 + m;
  const float br = bih[col] + bhh[col];
  const float bz = bih[HDIM + col] + bhh[HDIM + col];
  const float bni = bih[2 * HDIM + col];
  const float bnh = bhh[2 * HDIM + col];
#pragma unroll
  for (int v = 0; v < 8; ++v) {
    const float r = 1.0f / (1.0f + __expf(-(cr[v] + br)));
    const float z = 1.0f / (1.0f + __expf(-(cz[v] + bz)));
    const float nn = tanhf(cn[v] + bni + r * bnh);
    const int row = row0 + v + 8 * sel;
    Hout[(size_t)row * HDIM + col] = f2bf((1.0f - z) * nn);
  }
}

// ---------------------------------------------------------------------------
extern "C" void kernel_launch(void* const* d_in, const int* in_sizes, int n_in,
                              void* d_out, int out_size, void* d_ws, size_t ws_size,
                              hipStream_t stream) {
  const float* inputs = (const float*)d_in[0];
  const float* w_in   = (const float*)d_in[1];
  const float* b_in   = (const float*)d_in[2];
  const float* W_ih   = (const float*)d_in[3];
  // d_in[4] = W_hh: unused — h_prev == 0 so h@Whh^T == 0; only b_hh survives.
  const float* b_ih   = (const float*)d_in[5];
  const float* b_hh   = (const float*)d_in[6];
  const float* w_out  = (const float*)d_in[7];
  const float* b_out  = (const float*)d_in[8];
  float* out = (float*)d_out;

  char* ws = (char*)d_ws;
  bf16* wInT  = (bf16*)ws; ws += (size_t)NGROUP * GSZ * GSZ * sizeof(bf16);
  bf16* wOutT = (bf16*)ws; ws += (size_t)NGROUP * GSZ * GSZ * sizeof(bf16);
  bf16* WihB  = (bf16*)ws; ws += (size_t)2 * GATES * HDIM * sizeof(bf16);
  bf16* X1    = (bf16*)ws; ws += (size_t)BATCH * HDIM * sizeof(bf16);
  bf16* X2    = (bf16*)ws;

  {
    const int total = 2 * NGROUP * GSZ * GSZ + 2 * GATES * HDIM;
    k_convert<<<(total + 255) / 256, 256, 0, stream>>>(w_in, W_ih, w_out,
                                                       wInT, wOutT, WihB);
  }

  const int tiles = (BATCH / 16) * (HDIM / 16);   // 32768 wave tiles
  dim3 blk(128);                                   // 4 waves (wave32)
  dim3 grd(tiles / 4);

  k_grouped<float, bf16><<<grd, blk, 0, stream>>>(inputs, wInT, b_in, X1);
  k_gru<<<grd, blk, 0, stream>>>(X1, WihB,                       b_ih,        b_hh,        X2);
  k_gru<<<grd, blk, 0, stream>>>(X2, WihB + (size_t)GATES * HDIM, b_ih + GATES, b_hh + GATES, X1);
  k_grouped<bf16, float><<<grd, blk, 0, stream>>>(X1, wOutT, b_out, out);
}
